// Model_74612171866657
// MI455X (gfx1250) — matrix operand order repaired and checked by reading, not compiled, not run on hardware
//
#include <hip/hip_runtime.h>
#include <hip/hip_bf16.h>
#include <stdint.h>

// ---------------------------------------------------------------------------
// Types for CDNA5 WMMA / TDM builtins
// ---------------------------------------------------------------------------
typedef __attribute__((ext_vector_type(16))) __bf16 v16bf;
typedef __attribute__((ext_vector_type(8)))  float  v8f;
typedef __attribute__((ext_vector_type(4)))  unsigned int u32x4;
typedef __attribute__((ext_vector_type(8)))  int    i32x8;
typedef __attribute__((ext_vector_type(4)))  int    i32x4;

#if defined(__has_builtin)
#if __has_builtin(__builtin_amdgcn_tensor_load_to_lds) && __has_builtin(__builtin_amdgcn_s_wait_tensorcnt)
#define USE_TDM 1
#endif
#endif
#ifndef USE_TDM
#define USE_TDM 0
#endif

// Hardware bf16 convert (backend picks v_cvt_*bf16*; avoids manual RNE bit ops)
__device__ __forceinline__ __bf16 f2bf(float f) { return (__bf16)f; }

#if USE_TDM
// Build a D# for a 2D tile (tile1 rows x tile0 elements, fp32) and issue TDM load.
// OOB (tensor_dim clamp) returns zeros; optional LDS padding for bank-conflict-free reads.
__device__ __forceinline__ void tdm_load_2d(unsigned int ldsoff, const float* g,
                                            unsigned int tile0, unsigned int tile1,
                                            long tensor0, long tensor1, long stride0,
                                            int pad_interval_code, int pad_amount_code,
                                            int pad_en) {
  unsigned long long ga = (unsigned long long)(const void*)g;
  u32x4 g0;
  g0[0] = 1u;                                              // count=1, user mode
  g0[1] = ldsoff;                                          // lds_addr
  g0[2] = (unsigned int)(ga & 0xFFFFFFFFull);              // global_addr[31:0]
  g0[3] = (unsigned int)((ga >> 32) & 0x1FFFFFFull) | (2u << 30); // addr[56:32], type=2
  unsigned int t0 = (unsigned int)(tensor0 < 0 ? 0 : tensor0);
  unsigned int t1 = (unsigned int)(tensor1 < 0 ? 0 : tensor1);
  unsigned long long s0 = (unsigned long long)stride0;
  i32x8 g1;
  g1[0] = (int)((2u << 16) |                               // data_size = 4B
                ((unsigned)(pad_en ? 1 : 0) << 20) |
                ((unsigned)pad_interval_code << 22) |
                ((unsigned)pad_amount_code << 25));
  g1[1] = (int)((t0 & 0xFFFFu) << 16);                     // tensor_dim0 lo16 at bit48
  g1[2] = (int)((t0 >> 16) | ((t1 & 0xFFFFu) << 16));      // dim0 hi16 | dim1 lo16
  g1[3] = (int)((t1 >> 16) | (tile0 << 16));               // dim1 hi16 | tile_dim0
  g1[4] = (int)(tile1 & 0xFFFFu);                          // tile_dim1, tile_dim2=0
  g1[5] = (int)(unsigned int)(s0 & 0xFFFFFFFFull);         // tensor_dim0_stride lo32
  g1[6] = (int)(unsigned int)((s0 >> 32) & 0xFFFFull);     // stride hi16
  g1[7] = 0;
  i32x4 z4 = {0, 0, 0, 0};
#if defined(__clang_major__) && __clang_major__ >= 23
  i32x8 z8 = {0, 0, 0, 0, 0, 0, 0, 0};
  __builtin_amdgcn_tensor_load_to_lds(g0, g1, z4, z4, z8, 0);
#else
  __builtin_amdgcn_tensor_load_to_lds(g0, g1, z4, z4, 0);
#endif
}
#endif

// ---------------------------------------------------------------------------
// Batched WMMA GEMM: C[z] = act(A[z](MxK) * B[z](KxN) + biasN + biasM) (+resid)
// A row-major lda, B row-major ldb, C row-major ldc. amod>0: A batch = z%amod,
// biasM batch offset = (z%amod)*M. act: 0=none 1=relu 2=softplus
// Double-buffered LDS; on the CDNA5 path the next tile pair is DMA'd by the TDM
// while the current pair feeds the WMMAs (wait tensorcnt<=2 = older pair done).
// ---------------------------------------------------------------------------
#define BM 64
#define BN 128
#define BK 32

__global__ __launch_bounds__(256) void k_gemm(
    const float* __restrict__ A, const float* __restrict__ B,
    const float* __restrict__ biasN, const float* __restrict__ biasM,
    const float* __restrict__ resid, float* __restrict__ C,
    int M, int N, int K, long lda, long ldb, long ldc,
    long sA, long sB, long sC, int amod, int act) {
  int z = blockIdx.z;
  long za = (amod > 0) ? (long)(z % amod) : (long)z;
  A += za * sA;
  B += (long)z * sB;
  C += (long)z * sC;
  if (resid) resid += (long)z * sC;
  if (biasM) biasM += za * (long)M;

  __shared__ float As[2][BM][BK + 1];  // 33-dword row stride: conflict-free A frags
  __shared__ float Bs[2][BK][BN];

  int m0 = blockIdx.y * BM;
  int n0 = blockIdx.x * BN;
  int wave = threadIdx.x >> 5;
  int lane = threadIdx.x & 31;
  int wm = (wave >> 2) * 32;      // 0 / 32
  int wn = (wave & 3) * 32;       // 0 / 32 / 64 / 96
  int ln16 = lane & 15;
  int hi = lane >> 4;

  v8f acc[2][2] = {};

#if USE_TDM
  // Prologue: DMA first tile pair into buffer 0
  if (threadIdx.x < 32) {
    unsigned int ldsA0 = (unsigned int)(unsigned long long)(const void*)&As[0][0][0];
    unsigned int ldsB0 = (unsigned int)(unsigned long long)(const void*)&Bs[0][0][0];
    tdm_load_2d(ldsA0, A + (long)m0 * lda, BK, BM,
                (long)K, (long)M - m0, lda, /*interval 32dw*/ 4, /*amount 1dw*/ 0, 1);
    tdm_load_2d(ldsB0, B + n0, BN, BK,
                (long)N - n0, (long)K, ldb, 0, 0, 0);
  }
#else
  for (int i = threadIdx.x; i < BM * BK; i += 256) {
    int r = i >> 5, c = i & 31;
    long gr = m0 + r, gc = c;
    As[0][r][c] = (gr < M && gc < K) ? A[gr * lda + gc] : 0.f;
  }
  for (int i = threadIdx.x; i < BK * BN; i += 256) {
    int r = i >> 7, c = i & 127;
    long gr = r, gc = n0 + c;
    Bs[0][r][c] = (gr < K && gc < N) ? B[gr * ldb + gc] : 0.f;
  }
#endif

  int cur = 0;
  for (int k0 = 0; k0 < K; k0 += BK, cur ^= 1) {
    int knext = k0 + BK;
#if USE_TDM
    if (threadIdx.x < 32) {
      if (knext < K) {
        unsigned int ldsA = (unsigned int)(unsigned long long)(const void*)&As[cur ^ 1][0][0];
        unsigned int ldsB = (unsigned int)(unsigned long long)(const void*)&Bs[cur ^ 1][0][0];
        tdm_load_2d(ldsA, A + (long)m0 * lda + knext, BK, BM,
                    (long)K - knext, (long)M - m0, lda, 4, 0, 1);
        tdm_load_2d(ldsB, B + (long)knext * ldb + n0, BN, BK,
                    (long)N - n0, (long)K - knext, ldb, 0, 0, 0);
        __builtin_amdgcn_s_wait_tensorcnt(2);  // current pair done; next pair in flight
      } else {
        __builtin_amdgcn_s_wait_tensorcnt(0);
      }
    }
    __syncthreads();
#else
    if (knext < K) {
      for (int i = threadIdx.x; i < BM * BK; i += 256) {
        int r = i >> 5, c = i & 31;
        long gr = m0 + r, gc = knext + c;
        As[cur ^ 1][r][c] = (gr < M && gc < K) ? A[gr * lda + gc] : 0.f;
      }
      for (int i = threadIdx.x; i < BK * BN; i += 256) {
        int r = i >> 7, c = i & 127;
        long gr = knext + r, gc = n0 + c;
        Bs[cur ^ 1][r][c] = (gr < K && gc < N) ? B[gr * ldb + gc] : 0.f;
      }
    }
    __syncthreads();
#endif
    // Build fragments (ISA 7.12.2 16-bit layouts) and issue 4 WMMAs
    v16bf af[2], bf[2];
    int koff = hi * 8;
#pragma unroll
    for (int i = 0; i < 2; ++i) {
      const float* Ap = &As[cur][wm + i * 16 + ln16][0];
#pragma unroll
      for (int t = 0; t < 8; ++t) af[i][t] = f2bf(Ap[koff + t]);
#pragma unroll
      for (int t = 0; t < 8; ++t) af[i][8 + t] = f2bf(Ap[16 + koff + t]);
    }
    int kb = hi * 16;
#pragma unroll
    for (int j = 0; j < 2; ++j) {
      int col = wn + j * 16 + ln16;
#pragma unroll
      for (int t = 0; t < 16; ++t) bf[j][t] = f2bf(Bs[cur][(kb >> 1) + t + (t & 8)][col]);
    }
#pragma unroll
    for (int i = 0; i < 2; ++i)
#pragma unroll
      for (int j = 0; j < 2; ++j)
        acc[i][j] = __builtin_amdgcn_wmma_f32_16x16x32_bf16(
            false, af[i], false, bf[j], (short)0, acc[i][j], false, false);
    __syncthreads();  // all waves done reading buf[cur]; next iter may DMA into it
  }

#pragma unroll
  for (int i = 0; i < 2; ++i)
#pragma unroll
    for (int j = 0; j < 2; ++j) {
      int grb = m0 + wm + i * 16 + hi * 8;
      int gc = n0 + wn + j * 16 + ln16;
      if (gc >= N) continue;
#pragma unroll
      for (int r = 0; r < 8; ++r) {
        int gr = grb + r;
        if (gr >= M) continue;
        float v = acc[i][j][r];
        if (biasN) v += biasN[gc];
        if (biasM) v += biasM[gr];
        if (act == 1) v = v > 0.f ? v : 0.f;
        else if (act == 2) v = (v > 20.f) ? v : log1pf(expf(v));
        if (resid) v += resid[(long)gr * ldc + gc];
        C[(long)gr * ldc + gc] = v;
      }
    }
}

// ---------------------------------------------------------------------------
// Patch embed: 4x4 s4 conv (NCHW in) + bias + LN -> (B,56,56,96) NHWC fp32
// ---------------------------------------------------------------------------
__global__ __launch_bounds__(128) void k_patch_embed(
    const float* __restrict__ x, const float* __restrict__ w,
    const float* __restrict__ bias, const float* __restrict__ g,
    const float* __restrict__ be, float* __restrict__ out) {
  int pix = blockIdx.x;           // b*3136 + oh*56 + ow
  int b = pix / 3136;
  int rem = pix % 3136;
  int oh = rem / 56, ow = rem % 56;
  int t = threadIdx.x;
  __shared__ float v[96];
  __shared__ float mmv[2];
  if (t < 96) {
    float acc = bias[t];
    const float* wp = w + t * 48;
#pragma unroll
    for (int c = 0; c < 3; ++c)
      for (int kh = 0; kh < 4; ++kh)
        for (int kw = 0; kw < 4; ++kw)
          acc += x[(((long)b * 3 + c) * 224 + oh * 4 + kh) * 224 + ow * 4 + kw] *
                 wp[(c * 4 + kh) * 4 + kw];
    v[t] = acc;
  }
  __syncthreads();
  if (t == 0) {
    float s = 0.f, ss = 0.f;
    for (int c = 0; c < 96; ++c) { s += v[c]; ss += v[c] * v[c]; }
    float m = s / 96.f;
    mmv[0] = m;
    mmv[1] = rsqrtf(ss / 96.f - m * m + 1e-5f);
  }
  __syncthreads();
  if (t < 96) out[(long)pix * 96 + t] = (v[t] - mmv[0]) * mmv[1] * g[t] + be[t];
}

// ---------------------------------------------------------------------------
// Row LayerNorm over last dim C (one block per row)
// ---------------------------------------------------------------------------
__global__ __launch_bounds__(256) void k_layernorm(
    const float* __restrict__ x, const float* __restrict__ g,
    const float* __restrict__ b, float* __restrict__ out, int C) {
  long row = blockIdx.x;
  const float* xr = x + row * C;
  float s = 0.f, ss = 0.f;
  for (int c = threadIdx.x; c < C; c += 256) { float v = xr[c]; s += v; ss += v * v; }
  __shared__ float rs[256], rq[256];
  rs[threadIdx.x] = s; rq[threadIdx.x] = ss;
  __syncthreads();
  for (int off = 128; off > 0; off >>= 1) {
    if (threadIdx.x < off) { rs[threadIdx.x] += rs[threadIdx.x + off]; rq[threadIdx.x] += rq[threadIdx.x + off]; }
    __syncthreads();
  }
  float mean = rs[0] / C;
  float inv = rsqrtf(rq[0] / C - mean * mean + 1e-5f);
  float* orow = out + row * C;
  for (int c = threadIdx.x; c < C; c += 256)
    orow[c] = (xr[c] - mean) * inv * g[c] + b[c];
}

// ---------------------------------------------------------------------------
// Depthwise 3x3 SAME conv + bias + SiLU. in: xz (B,L,4C) first di chans.
// out xc: (B,di,L) channel-major for the scan.
// ---------------------------------------------------------------------------
__global__ void k_dwconv(const float* __restrict__ xz, const float* __restrict__ cw,
                         const float* __restrict__ cb, float* __restrict__ xc,
                         int Bn, int H, int W, int di, int C4) {
  long idx = (long)blockIdx.x * blockDim.x + threadIdx.x;
  long total = (long)Bn * di * H * W;
  if (idx >= total) return;
  int w = (int)(idx % W); long r = idx / W;
  int h = (int)(r % H); r /= H;
  int d = (int)(r % di); int b = (int)(r / di);
  const float* wp = cw + (long)d * 9;
  float acc = cb[d];
#pragma unroll
  for (int kh = -1; kh <= 1; ++kh)
#pragma unroll
    for (int kw = -1; kw <= 1; ++kw) {
      int hh = h + kh, ww = w + kw;
      if (hh >= 0 && hh < H && ww >= 0 && ww < W)
        acc += xz[((long)(b * H + hh) * W + ww) * C4 + d] * wp[(kh + 1) * 3 + (kw + 1)];
    }
  float sv = acc / (1.f + __expf(-acc));
  xc[((long)b * di + d) * (long)(H * W) + (long)h * W + w] = sv;
}

// 4-direction gather: xs (B,4,di,L)
__global__ void k_build_xs(const float* __restrict__ xc, float* __restrict__ xs,
                           int Bn, int di, int H, int W) {
  long L = (long)H * W;
  long idx = (long)blockIdx.x * blockDim.x + threadIdx.x;
  long total = (long)Bn * 4 * di * L;
  if (idx >= total) return;
  long l = idx % L; long r = idx / L;
  int d = (int)(r % di); r /= di;
  int k = (int)(r % 4); int b = (int)(r / 4);
  const float* src = xc + ((long)b * di + d) * L;
  float v;
  if (k == 0) v = src[l];
  else if (k == 1) { int w = (int)(l / H), h = (int)(l % H); v = src[(long)h * W + w]; }
  else if (k == 2) v = src[L - 1 - l];
  else { long lr = L - 1 - l; int w = (int)(lr / H), h = (int)(lr % H); v = src[(long)h * W + w]; }
  xs[idx] = v;
}

// ---------------------------------------------------------------------------
// Selective scan: one thread per (b,dir,channel); h[16] in VGPRs, L serial.
// ---------------------------------------------------------------------------
__global__ void k_scan(const float* __restrict__ xs, const float* __restrict__ dt,
                       const float* __restrict__ xdbl, const float* __restrict__ A_log,
                       const float* __restrict__ Dp, float* __restrict__ ys,
                       int Bn, int di, long L, int R, int R2N) {
  long idx = (long)blockIdx.x * blockDim.x + threadIdx.x;
  long total = (long)Bn * 4 * di;
  if (idx >= total) return;
  int d = (int)(idx % di); long r = idx / di;
  int k = (int)(r % 4); int b = (int)(r / 4);
  float Av[16];
#pragma unroll
  for (int n = 0; n < 16; ++n) Av[n] = -__expf(A_log[((long)k * di + d) * 16 + n]);
  float Dv = Dp[(long)k * di + d];
  const float* up = xs + idx * L;
  const float* dtp = dt + idx * L;
  const float* Bp = xdbl + ((long)(b * 4 + k) * R2N + R) * L;
  const float* Cp = Bp + 16 * L;
  float* yp = ys + idx * L;
  float h[16];
#pragma unroll
  for (int n = 0; n < 16; ++n) h[n] = 0.f;
  for (long t = 0; t < L; ++t) {
    float u = up[t], dtv = dtp[t];
    float du = dtv * u;
    float y = 0.f;
#pragma unroll
    for (int n = 0; n < 16; ++n) {
      float dA = __expf(dtv * Av[n]);
      h[n] = dA * h[n] + du * Bp[n * L + t];
      y += h[n] * Cp[n * L + t];
    }
    yp[t] = y + Dv * u;
  }
}

// ---------------------------------------------------------------------------
// Merge 4 directions + out-LN + SiLU(z) gate -> ymg (B*L, di)
// ---------------------------------------------------------------------------
__global__ __launch_bounds__(256) void k_merge(
    const float* __restrict__ ys, const float* __restrict__ xz,
    const float* __restrict__ g, const float* __restrict__ bta,
    float* __restrict__ ymg, int H, int W, int di, int C4) {
  long L = (long)H * W;
  long bl = blockIdx.x;                 // b*L + l
  int b = (int)(bl / L);
  long l = bl % L;
  int hh = (int)(l / W), ww = (int)(l % W);
  long lw = (long)ww * H + hh;
  const float* y0 = ys + ((long)b * 4 + 0) * di * L;
  const float* y1 = ys + ((long)b * 4 + 1) * di * L;
  const float* y2 = ys + ((long)b * 4 + 2) * di * L;
  const float* y3 = ys + ((long)b * 4 + 3) * di * L;
  __shared__ float vbuf[1536];
  __shared__ float rs[256], rq[256];
  float s = 0.f, ss = 0.f;
  for (int d = threadIdx.x; d < di; d += 256) {
    float v = y0[(long)d * L + l] + y2[(long)d * L + (L - 1 - l)] +
              y1[(long)d * L + lw] + y3[(long)d * L + (L - 1 - lw)];
    vbuf[d] = v; s += v; ss += v * v;
  }
  rs[threadIdx.x] = s; rq[threadIdx.x] = ss;
  __syncthreads();
  for (int off = 128; off > 0; off >>= 1) {
    if (threadIdx.x < off) { rs[threadIdx.x] += rs[threadIdx.x + off]; rq[threadIdx.x] += rq[threadIdx.x + off]; }
    __syncthreads();
  }
  float mean = rs[0] / di;
  float inv = rsqrtf(rq[0] / di - mean * mean + 1e-5f);
  for (int d = threadIdx.x; d < di; d += 256) {
    float v = (vbuf[d] - mean) * inv * g[d] + bta[d];
    float zv = xz[bl * C4 + di + d];
    ymg[bl * di + d] = v * (zv / (1.f + __expf(-zv)));
  }
}

// Patch-merging gather -> (B, H/2, W/2, 4C)
__global__ void k_dsgather(const float* __restrict__ h, float* __restrict__ out,
                           int Bn, int H, int W, int C) {
  int H2 = H / 2, W2 = W / 2;
  long idx = (long)blockIdx.x * blockDim.x + threadIdx.x;
  long total = (long)Bn * H2 * W2 * 4 * C;
  if (idx >= total) return;
  int c4 = (int)(idx % (4 * C)); long r = idx / (4 * C);
  int j = (int)(r % W2); r /= W2;
  int i = (int)(r % H2); int b = (int)(r / H2);
  int q = c4 / C, c = c4 % C;
  int hi_ = 2 * i + ((q == 1 || q == 3) ? 1 : 0);
  int wi = 2 * j + ((q >= 2) ? 1 : 0);
  out[idx] = h[(((long)b * H + hi_) * W + wi) * C + c];
}

__global__ void k_meanpool(const float* __restrict__ x, float* __restrict__ out,
                           int Lp, int C) {
  int idx = blockIdx.x * blockDim.x + threadIdx.x;
  if (idx >= 4 * C) return;
  int b = idx / C, c = idx % C;
  float s = 0.f;
  for (int l = 0; l < Lp; ++l) s += x[((long)b * Lp + l) * C + c];
  out[idx] = s / Lp;
}

// ---------------------------------------------------------------------------
// Host orchestration
// ---------------------------------------------------------------------------
extern "C" void kernel_launch(void* const* d_in, const int* in_sizes, int n_in,
                              void* d_out, int out_size, void* d_ws, size_t ws_size,
                              hipStream_t stream) {
  (void)in_sizes; (void)n_in; (void)out_size; (void)ws_size;
  int idx = 0;
  auto nxt = [&]() -> const float* { return (const float*)d_in[idx++]; };

  const float* x = nxt();
  const float* pe_w = nxt(); const float* pe_b = nxt();
  const float* pe_g = nxt(); const float* pe_be = nxt();

  struct Blk {
    const float *ln_g, *ln_b, *in_w, *in_b, *conv_w, *conv_b, *xproj_w,
        *dtproj_w, *dtproj_b, *A_log, *Dp, *outln_g, *outln_b, *out_w, *out_b;
  };
  Blk blks[15]; int nb = 0;
  const float *dsg[3], *dsb[3], *dsw[3];
  const int depths[4] = {2, 2, 9, 2};
  for (int s = 0; s < 4; ++s) {
    for (int d2 = 0; d2 < depths[s]; ++d2) {
      Blk& B = blks[nb++];
      B.ln_g = nxt(); B.ln_b = nxt(); B.in_w = nxt(); B.in_b = nxt();
      B.conv_w = nxt(); B.conv_b = nxt(); B.xproj_w = nxt();
      B.dtproj_w = nxt(); B.dtproj_b = nxt(); B.A_log = nxt(); B.Dp = nxt();
      B.outln_g = nxt(); B.outln_b = nxt(); B.out_w = nxt(); B.out_b = nxt();
    }
    if (s < 3) { dsg[s] = nxt(); dsb[s] = nxt(); dsw[s] = nxt(); }
  }
  const float* hd_g = nxt(); const float* hd_b = nxt();
  const float* hd_w = nxt(); const float* hd_bb = nxt();
  const float* fl_w = nxt(); const float* fl_b = nxt();
  const float* fr_w = nxt(); const float* fr_b = nxt();

  // Workspace carve-up (fp32 elements)
  float* ws = (float*)d_ws;
  float* hA = ws;                       // B*L*C max      = 1,204,224
  float* hB = hA + 1204224;
  float* xzb = hB + 1204224;            // B*L*4C max     = 4,816,896
  float* xcb = xzb + 4816896;           // B*di*L max     = 2,408,448
  float* xsb = xcb + 2408448;           // B*4*di*L max   = 9,633,792
  float* xdb = xsb + 9633792;           // B*4*R2N*L max  < 4,014,080
  float* dtb = xdb + 4014080;           // 9,633,792
  float* ysb = dtb + 9633792;           // 9,633,792
  float* ymg = ysb + 9633792;           // 2,408,448
  float* pooled = ymg + 2408448;        // 3072
  float* logits = pooled + 4 * 768;     // 4000

  auto gemm = [&](const float* A, const float* Bm, const float* biasN,
                  const float* biasM, const float* resid, float* Cp,
                  int M, int N, int K, long lda, long ldb, long ldc,
                  long sA, long sB, long sC, int nbatch, int amod, int act) {
    dim3 g((N + BN - 1) / BN, (M + BM - 1) / BM, nbatch);
    k_gemm<<<g, dim3(256), 0, stream>>>(A, Bm, biasN, biasM, resid, Cp, M, N, K,
                                        lda, ldb, ldc, sA, sB, sC, amod, act);
  };

  k_patch_embed<<<4 * 56 * 56, 128, 0, stream>>>(x, pe_w, pe_b, pe_g, pe_be, hA);

  int C = 96, H = 56, W = 56, bi = 0;
  for (int s = 0; s < 4; ++s) {
    int L = H * W, di = 2 * C, C4 = 4 * C, R = C / 16, R2N = R + 32;
    for (int d2 = 0; d2 < depths[s]; ++d2) {
      const Blk& B = blks[bi++];
      k_layernorm<<<4 * L, 256, 0, stream>>>(hA, B.ln_g, B.ln_b, hB, C);
      // in_proj: (4L,C) x (C,4C)
      gemm(hB, B.in_w, B.in_b, nullptr, nullptr, xzb, 4 * L, C4, C,
           C, C4, C4, 0, 0, 0, 1, 0, 0);
      {
        long tot = (long)4 * di * L;
        k_dwconv<<<(int)((tot + 255) / 256), 256, 0, stream>>>(
            xzb, B.conv_w, B.conv_b, xcb, 4, H, W, di, C4);
      }
      {
        long tot = (long)4 * 4 * di * L;
        k_build_xs<<<(int)((tot + 255) / 256), 256, 0, stream>>>(xcb, xsb, 4, di, H, W);
      }
      // x_dbl: per (b,k): (R2N,di) x (di,L)
      gemm(B.xproj_w, xsb, nullptr, nullptr, nullptr, xdb, R2N, L, di,
           di, L, L, (long)R2N * di, (long)di * L, (long)R2N * L, 16, 4, 0);
      // dt: per (b,k): (di,R) x (R,L) + dtproj_b[row], softplus
      gemm(B.dtproj_w, xdb, nullptr, B.dtproj_b, nullptr, dtb, di, L, R,
           R, L, L, (long)di * R, (long)R2N * L, (long)di * L, 16, 4, 2);
      {
        int tot = 4 * 4 * di;
        k_scan<<<(tot + 255) / 256, 256, 0, stream>>>(
            xsb, dtb, xdb, B.A_log, B.Dp, ysb, 4, di, (long)L, R, R2N);
      }
      k_merge<<<4 * L, 256, 0, stream>>>(ysb, xzb, B.outln_g, B.outln_b, ymg, H, W, di, C4);
      // out_proj + residual add into hA
      gemm(ymg, B.out_w, B.out_b, nullptr, hA, hA, 4 * L, C, di,
           di, C, C, 0, 0, 0, 1, 0, 0);
    }
    if (s < 3) {
      {
        long tot = (long)4 * L * C;
        k_dsgather<<<(int)((tot + 255) / 256), 256, 0, stream>>>(hA, hB, 4, H, W, C);
      }
      k_layernorm<<<4 * (L / 4), 256, 0, stream>>>(hB, dsg[s], dsb[s], hB, 4 * C);
      gemm(hB, dsw[s], nullptr, nullptr, nullptr, hA, 4 * (L / 4), 2 * C, 4 * C,
           4 * C, 2 * C, 2 * C, 0, 0, 0, 1, 0, 0);
      C *= 2; H /= 2; W /= 2;
    }
  }

  // Head: LN -> mean pool -> (4,768)x(768,1000)+b, relu -> fcL / fcR
  k_layernorm<<<4 * 49, 256, 0, stream>>>(hA, hd_g, hd_b, hB, 768);
  k_meanpool<<<(4 * 768 + 255) / 256, 256, 0, stream>>>(hB, pooled, 49, 768);
  gemm(pooled, hd_w, hd_bb, nullptr, nullptr, logits, 4, 1000, 768,
       768, 1000, 1000, 0, 0, 0, 1, 0, 1);
  float* outp = (float*)d_out;
  gemm(logits, fl_w, fl_b, nullptr, nullptr, outp, 4, 19004, 1000,
       1000, 19004, 19004, 0, 0, 0, 1, 0, 0);
  gemm(logits, fr_w, fr_b, nullptr, nullptr, outp + 4 * 19004, 4, 20544, 1000,
       1000, 20544, 20544, 0, 0, 0, 1, 0, 0);
}
